// DoRACrossAttention_78623671320686
// MI455X (gfx1250) — matrix operand order repaired and checked by reading, not compiled, not run on hardware
//
#include <hip/hip_runtime.h>

typedef __attribute__((ext_vector_type(16))) __bf16 v16bf;
typedef __attribute__((ext_vector_type(8)))  __bf16 v8bf;
typedef __attribute__((ext_vector_type(8)))  float  v8f;

#define B_   4
#define N_   2048
#define C_   1024
#define H_   16
#define D_   64
#define R_   16
#define BN_  8192      // B_*N_
#define SCALE_ 0.125f  // D^-0.5

// ---------------- WMMA helpers ----------------

__device__ __forceinline__ v8f wmma_bf16(v16bf a, v16bf b, v8f c) {
  // emits v_wmma_f32_16x16x32_bf16
  return __builtin_amdgcn_wmma_f32_16x16x32_bf16(false, a, false, b, (short)0, c,
                                                 false, false);
}

__device__ __forceinline__ v8f vzero8() {
  v8f z;
#pragma unroll
  for (int i = 0; i < 8; ++i) z[i] = 0.f;
  return z;
}

// A fragment (16x32 bf16): caller passes rowptr + k0 + (lane>>4)*8.
// elems 0..7 = K base..base+7, elems 8..15 = K base+16..base+23
__device__ __forceinline__ v16bf load_frag_a(const __bf16* p) {
  v8bf lo = *(const v8bf*)(p);
  v8bf hi = *(const v8bf*)(p + 16);
  v16bf r;
#pragma unroll
  for (int i = 0; i < 8; ++i) { r[i] = lo[i]; r[i + 8] = hi[i]; }
  return r;
}

// B fragment (32x16 bf16): caller passes colptr + k0 + (lane>>4)*8 ;
// elems 0..7 = K base..base+7, elems 8..15 = K base+16..base+23
__device__ __forceinline__ v16bf load_frag_b(const __bf16* p) {
  return load_frag_a(p);
}

// ---------------- Kernel 1: fp32 -> bf16 converts (+ weight transposes) ----

__global__ __launch_bounds__(256) void k_convert(const float* __restrict__ q,
                                                 const float* __restrict__ wqkv,
                                                 const float* __restrict__ wproj,
                                                 __bf16* __restrict__ qbf,
                                                 __bf16* __restrict__ wqkvT,
                                                 __bf16* __restrict__ wprojT) {
  const int NQ = BN_ * C_;       // 8388608
  const int NW = 3 * C_ * C_;    // 3145728
  int i = blockIdx.x * 256 + threadIdx.x;
  if (i < NQ) {
    qbf[i] = (__bf16)q[i];
  } else if (i < NQ + NW) {
    int t = i - NQ;
    int n = t >> 10, k = t & 1023;           // wqkvT[n][k] = W_qkv[k][n]
    wqkvT[t] = (__bf16)wqkv[k * 3072 + n];
  } else {
    int t = i - NQ - NW;
    int n = t >> 10, k = t & 1023;           // wprojT[n][k] = W_proj[k][n]
    wprojT[t] = (__bf16)wproj[k * 1024 + n];
  }
}

// ---------------- Kernel 2: LoRA tmp = query @ W_loraA   [BN,16] ----------

__global__ __launch_bounds__(256) void k_lora(const float* __restrict__ q,
                                              const float* __restrict__ wA,
                                              float* __restrict__ tmp) {
  int gid = blockIdx.x * 256 + threadIdx.x;  // 131072
  int row = gid >> 4, r = gid & 15;
  const float* qr = q + row * C_;
  float s = 0.f;
  for (int k = 0; k < C_; ++k) s += qr[k] * wA[k * R_ + r];
  tmp[gid] = s;
}

// ---------------- Kernel 3: QKV GEMM (bf16 WMMA, ping-pong pipelined) ------
// Each wave: 16 rows x 64 cols. Scatter into q(f32)[B,H,N,D], k(bf16)[B,H,N,D],
// v(bf16)[B,H,D,N] (transposed for the PV matmul).

__global__ __launch_bounds__(256) void k_gemm_qkv(const __bf16* __restrict__ A,
                                                  const __bf16* __restrict__ Bt,
                                                  float* __restrict__ qf,
                                                  __bf16* __restrict__ kbf,
                                                  __bf16* __restrict__ vbf) {
  int wave = (blockIdx.x * 256 + threadIdx.x) >> 5;
  int lane = threadIdx.x & 31;
  int mt = wave / 48, cb = wave % 48;        // 512 M-tiles x 48 col-blocks
  int m0 = mt * 16;
  int row = lane & 15;
  int ka = (lane >> 4) * 8, kb16 = (lane >> 4) * 8, mi = (lane >> 4) * 8;

  v8f acc[4];
#pragma unroll
  for (int j = 0; j < 4; ++j) acc[j] = vzero8();

  const __bf16* ap = A + (m0 + row) * C_ + ka;
  const __bf16* bp = Bt + (cb * 64 + row) * C_ + kb16;

  // ping-pong fragment sets: loads write straight into their consumers,
  // no rotation copies.
  v16bf af0 = load_frag_a(ap);
  v16bf bf0[4];
#pragma unroll
  for (int j = 0; j < 4; ++j) bf0[j] = load_frag_b(bp + j * 16 * C_);
  v16bf af1, bf1[4];

  for (int k0 = 0; k0 < C_ - 64; k0 += 64) {
    af1 = load_frag_a(ap + k0 + 32);
#pragma unroll
    for (int j = 0; j < 4; ++j) bf1[j] = load_frag_b(bp + j * 16 * C_ + k0 + 32);
#pragma unroll
    for (int j = 0; j < 4; ++j) acc[j] = wmma_bf16(af0, bf0[j], acc[j]);

    af0 = load_frag_a(ap + k0 + 64);
#pragma unroll
    for (int j = 0; j < 4; ++j) bf0[j] = load_frag_b(bp + j * 16 * C_ + k0 + 64);
#pragma unroll
    for (int j = 0; j < 4; ++j) acc[j] = wmma_bf16(af1, bf1[j], acc[j]);
  }
  af1 = load_frag_a(ap + C_ - 32);
#pragma unroll
  for (int j = 0; j < 4; ++j) bf1[j] = load_frag_b(bp + j * 16 * C_ + C_ - 32);
#pragma unroll
  for (int j = 0; j < 4; ++j) acc[j] = wmma_bf16(af0, bf0[j], acc[j]);
#pragma unroll
  for (int j = 0; j < 4; ++j) acc[j] = wmma_bf16(af1, bf1[j], acc[j]);

  int which = (cb * 64) >> 10;  // 0=q 1=k 2=v
  int h = cb & 15;
#pragma unroll
  for (int j = 0; j < 4; ++j) {
#pragma unroll
    for (int r = 0; r < 8; ++r) {
      int gr = m0 + r + mi;
      int b = gr >> 11, n = gr & 2047;
      int d = j * 16 + row;
      float v = acc[j][r];
      int idx = ((b * H_ + h) * N_ + n) * D_ + d;
      if (which == 0)      qf[idx] = v;
      else if (which == 1) kbf[idx] = (__bf16)v;
      else                 vbf[((b * H_ + h) * D_ + d) * N_ + n] = (__bf16)v;
    }
  }
}

// ---------------- Kernel 4: q += magnitude * (tmp @ W_loraB); -> bf16 ------

__global__ __launch_bounds__(256) void k_qfix(const float* __restrict__ qf,
                                              const float* __restrict__ tmp,
                                              const float* __restrict__ wB,
                                              const float* __restrict__ mag,
                                              __bf16* __restrict__ qb) {
  int gid = blockIdx.x * 256 + threadIdx.x;  // 8388608, [B,H,N,D] flat
  int d = gid & 63;
  int n = (gid >> 6) & 2047;
  int h = (gid >> 17) & 15;
  int b = gid >> 21;
  int rowi = b * N_ + n;
  float delta = 0.f;
#pragma unroll
  for (int r = 0; r < R_; ++r)
    delta += tmp[rowi * R_ + r] * wB[r * C_ + h * D_ + d];
  qb[gid] = (__bf16)(qf[gid] + mag[h] * delta);
}

// ---------------- Kernel 5: flash attention (one wave per 16 query rows) ---

__global__ __launch_bounds__(256) void k_attn(const __bf16* __restrict__ qb,
                                              const __bf16* __restrict__ kb,
                                              const __bf16* __restrict__ vb,
                                              const float* __restrict__ mask,
                                              __bf16* __restrict__ ob) {
  __shared__ __bf16 lds[8][16 * 64];
  int wid = threadIdx.x >> 5, lane = threadIdx.x & 31;
  int wave = blockIdx.x * 8 + wid;
  int bh = wave >> 7, qt = wave & 127;
  int b = bh >> 4, h = bh & 15;
  int q0 = qt * 16;

  const __bf16* qp = qb + (bh * N_ + q0) * D_;
  const __bf16* kp = kb + bh * N_ * D_;
  const __bf16* vp = vb + bh * D_ * N_;
  const float*  mp = mask + (b * N_ + q0) * N_;

  int row = lane & 15;
  int ka = (lane >> 4) * 8, kb16 = (lane >> 4) * 8, mi = (lane >> 4) * 8;

  v16bf qa[2];
#pragma unroll
  for (int ks = 0; ks < 2; ++ks)
    qa[ks] = load_frag_a(qp + row * D_ + ks * 32 + ka);

  v8f O[4];
  float mr[8], lr[8];
#pragma unroll
  for (int j = 0; j < 4; ++j) O[j] = vzero8();
#pragma unroll
  for (int r = 0; r < 8; ++r) { mr[r] = -1e30f; lr[r] = 0.f; }

  __bf16* pl = &lds[wid][0];

  for (int key0 = 0; key0 < N_; key0 += 64) {
    // ---- issue ALL loads for this key block up front (overlap with WMMAs) --
    v16bf kf[4][2];
#pragma unroll
    for (int j = 0; j < 4; ++j)
#pragma unroll
      for (int ks = 0; ks < 2; ++ks)
        kf[j][ks] = load_frag_b(kp + (key0 + j * 16 + row) * D_ + ks * 32 + kb16);

    float mv[4][8];
#pragma unroll
    for (int j = 0; j < 4; ++j)
#pragma unroll
      for (int r = 0; r < 8; ++r)
        mv[j][r] = mp[(r + mi) * N_ + key0 + j * 16 + row];

    v16bf vf[2][4];
#pragma unroll
    for (int ks = 0; ks < 2; ++ks)
#pragma unroll
      for (int j = 0; j < 4; ++j)
        vf[ks][j] = load_frag_b(vp + (j * 16 + row) * N_ + key0 + ks * 32 + kb16);

    // ---- S = q @ k^T ------------------------------------------------------
    v8f S[4];
#pragma unroll
    for (int j = 0; j < 4; ++j) S[j] = vzero8();
#pragma unroll
    for (int j = 0; j < 4; ++j)
#pragma unroll
      for (int ks = 0; ks < 2; ++ks)
        S[j] = wmma_bf16(qa[ks], kf[j][ks], S[j]);

    // ---- scale + mask, tile row max --------------------------------------
    float tm[8];
#pragma unroll
    for (int r = 0; r < 8; ++r) tm[r] = -1e30f;
#pragma unroll
    for (int j = 0; j < 4; ++j)
#pragma unroll
      for (int r = 0; r < 8; ++r) {
        float s = S[j][r] * SCALE_ + mv[j][r];
        S[j][r] = s;
        tm[r] = fmaxf(tm[r], s);
      }
    // row max across the 16 lanes of each half (rows live per half-wave)
#pragma unroll
    for (int off = 8; off >= 1; off >>= 1)
#pragma unroll
      for (int r = 0; r < 8; ++r)
        tm[r] = fmaxf(tm[r], __shfl_xor(tm[r], off, 32));
    // ---- online softmax update -------------------------------------------
    float al[8], ls[8];
#pragma unroll
    for (int r = 0; r < 8; ++r) {
      float mn = fmaxf(mr[r], tm[r]);
      al[r] = __expf(mr[r] - mn);
      mr[r] = mn;
      ls[r] = 0.f;
    }
#pragma unroll
    for (int j = 0; j < 4; ++j)
#pragma unroll
      for (int r = 0; r < 8; ++r) {
        float p = __expf(S[j][r] - mr[r]);
        ls[r] += p;
        pl[(r + mi) * 64 + j * 16 + row] = (__bf16)p;  // C-layout -> LDS tile
      }
#pragma unroll
    for (int off = 8; off >= 1; off >>= 1)
#pragma unroll
      for (int r = 0; r < 8; ++r) ls[r] += __shfl_xor(ls[r], off, 32);
#pragma unroll
    for (int r = 0; r < 8; ++r) lr[r] = lr[r] * al[r] + ls[r];
#pragma unroll
    for (int j = 0; j < 4; ++j)
#pragma unroll
      for (int r = 0; r < 8; ++r) O[j][r] *= al[r];

    asm volatile("s_wait_dscnt 0" ::: "memory");  // P tile visible wave-wide

    // ---- O += P @ V (A-frags from LDS, V-frags already resident) ----------
#pragma unroll
    for (int ks = 0; ks < 2; ++ks) {
      v16bf pf = load_frag_a(pl + row * 64 + ks * 32 + ka);
#pragma unroll
      for (int j = 0; j < 4; ++j)
        O[j] = wmma_bf16(pf, vf[ks][j], O[j]);
    }
    asm volatile("s_wait_dscnt 0" ::: "memory");  // reads done before next writes
  }

  // normalize and store as bf16 [B,N,H*D] for the output projection
#pragma unroll
  for (int j = 0; j < 4; ++j)
#pragma unroll
    for (int r = 0; r < 8; ++r) {
      float o = O[j][r] / lr[r];
      ob[(b * N_ + q0 + r + mi) * C_ + h * D_ + j * 16 + row] = (__bf16)o;
    }
}

// ---------------- Kernel 6: output projection GEMM (ping-pong pipelined) ---

__global__ __launch_bounds__(256) void k_gemm_proj(const __bf16* __restrict__ A,
                                                   const __bf16* __restrict__ Bt,
                                                   float* __restrict__ out) {
  int wave = (blockIdx.x * 256 + threadIdx.x) >> 5;
  int lane = threadIdx.x & 31;
  int mt = wave >> 4, cb = wave & 15;        // 512 M-tiles x 16 col-blocks
  int m0 = mt * 16;
  int row = lane & 15;
  int ka = (lane >> 4) * 8, kb16 = (lane >> 4) * 8, mi = (lane >> 4) * 8;

  v8f acc[4];
#pragma unroll
  for (int j = 0; j < 4; ++j) acc[j] = vzero8();

  const __bf16* ap = A + (m0 + row) * C_ + ka;
  const __bf16* bp = Bt + (cb * 64 + row) * C_ + kb16;

  v16bf af0 = load_frag_a(ap);
  v16bf bf0[4];
#pragma unroll
  for (int j = 0; j < 4; ++j) bf0[j] = load_frag_b(bp + j * 16 * C_);
  v16bf af1, bf1[4];

  for (int k0 = 0; k0 < C_ - 64; k0 += 64) {
    af1 = load_frag_a(ap + k0 + 32);
#pragma unroll
    for (int j = 0; j < 4; ++j) bf1[j] = load_frag_b(bp + j * 16 * C_ + k0 + 32);
#pragma unroll
    for (int j = 0; j < 4; ++j) acc[j] = wmma_bf16(af0, bf0[j], acc[j]);

    af0 = load_frag_a(ap + k0 + 64);
#pragma unroll
    for (int j = 0; j < 4; ++j) bf0[j] = load_frag_b(bp + j * 16 * C_ + k0 + 64);
#pragma unroll
    for (int j = 0; j < 4; ++j) acc[j] = wmma_bf16(af1, bf1[j], acc[j]);
  }
  af1 = load_frag_a(ap + C_ - 32);
#pragma unroll
  for (int j = 0; j < 4; ++j) bf1[j] = load_frag_b(bp + j * 16 * C_ + C_ - 32);
#pragma unroll
  for (int j = 0; j < 4; ++j) acc[j] = wmma_bf16(af0, bf0[j], acc[j]);
#pragma unroll
  for (int j = 0; j < 4; ++j) acc[j] = wmma_bf16(af1, bf1[j], acc[j]);

#pragma unroll
  for (int j = 0; j < 4; ++j)
#pragma unroll
    for (int r = 0; r < 8; ++r)
      out[(m0 + r + mi) * C_ + cb * 64 + j * 16 + row] = acc[j][r];
}

// ---------------- host launch ----------------

extern "C" void kernel_launch(void* const* d_in, const int* in_sizes, int n_in,
                              void* d_out, int out_size, void* d_ws, size_t ws_size,
                              hipStream_t stream) {
  (void)in_sizes; (void)n_in; (void)out_size; (void)ws_size;
  const float* query = (const float*)d_in[0];
  const float* mask  = (const float*)d_in[1];
  const float* wqkv  = (const float*)d_in[2];
  const float* mag   = (const float*)d_in[3];
  const float* wlA   = (const float*)d_in[4];
  const float* wlB   = (const float*)d_in[5];
  const float* wproj = (const float*)d_in[6];
  float* out = (float*)d_out;
  char* ws = (char*)d_ws;

  // workspace layout (bytes)
  __bf16* qbf    = (__bf16*)(ws + 0);          // 16 MB  query bf16 [BN,C]
  __bf16* wqkvT  = (__bf16*)(ws + 16777216);   //  6 MB  W_qkv^T bf16 [3C,C]
  __bf16* wprojT = (__bf16*)(ws + 23068672);   //  2 MB  W_proj^T bf16 [C,C]
  float*  ltmp   = (float*) (ws + 25165824);   // 0.5 MB lora tmp [BN,R]
  float*  qf32   = (float*) (ws + 25690112);   // 32 MB  q f32 [B,H,N,D]
  __bf16* kbf    = (__bf16*)(ws + 59244544);   // 16 MB  k bf16 [B,H,N,D]
  __bf16* vbf    = (__bf16*)(ws + 76021760);   // 16 MB  v bf16 [B,H,D,N]
  __bf16* qbhnd  = (__bf16*)(ws + 92798976);   // 16 MB  q+dora bf16 [B,H,N,D]
  __bf16* obf    = (__bf16*)(ws + 109576192);  // 16 MB  attn out bf16 [B,N,C]

  k_convert<<<49152, 256, 0, stream>>>(query, wqkv, wproj, qbf, wqkvT, wprojT);
  k_lora<<<512, 256, 0, stream>>>(query, wlA, ltmp);
  k_gemm_qkv<<<3072, 256, 0, stream>>>(qbf, wqkvT, qf32, kbf, vbf);
  k_qfix<<<32768, 256, 0, stream>>>(qf32, ltmp, wlB, mag, qbhnd);
  k_attn<<<1024, 256, 0, stream>>>(qbhnd, kbf, vbf, mask, obf);
  k_gemm_proj<<<1024, 256, 0, stream>>>(obf, wprojT, out);
}
